// Model_22926535426561
// MI455X (gfx1250) — matrix operand order repaired and checked by reading, not compiled, not run on hardware
//
#include <hip/hip_runtime.h>
#include <hip/hip_bf16.h>
#include <math.h>

typedef unsigned short u16;
typedef unsigned int   u32;
typedef __attribute__((ext_vector_type(16))) __bf16 v16bf;
typedef __attribute__((ext_vector_type(8)))  float  v8f;

#define SEQ  8192
#define EDIM 512
#define HDIM 1024
#define PITCH 40   // LDS panel pitch in halves: 80B rows, 16B aligned accesses

// ---------- bf16 helpers ----------
__device__ __forceinline__ u16 f2bf(float f) {
  u32 u = __builtin_bit_cast(u32, f);
  u32 r = u + 0x7FFFu + ((u >> 16) & 1u);   // round-to-nearest-even
  return (u16)(r >> 16);
}
__device__ __forceinline__ float bf2f(u16 h) {
  return __builtin_bit_cast(float, ((u32)h) << 16);
}

struct B32x2 { uint4 lo, hi; };

// A-fragment (16x32 bf16): lane<16 -> row=lane, K in {0..7,16..23};
// lane>=16 -> row=lane-16, K in {8..15,24..31}.  p pre-offset by K-half.
__device__ __forceinline__ v16bf load_a16(const u16* p) {
  B32x2 d;
  d.lo = *(const uint4*)(p);        // 8 halves: K khalf..khalf+7
  d.hi = *(const uint4*)(p + 16);   // 8 halves: K khalf+16..khalf+23
  return __builtin_bit_cast(v16bf, d);
}
// B-fragment (32x16 bf16, from W[N][K] row-major for A@W^T):
// lane<16 -> col=lane, K in {0..7,16..23}; lane>=16 -> K in {8..15,24..31}.
__device__ __forceinline__ v16bf load_b16(const u16* p) {
  B32x2 d;
  d.lo = *(const uint4*)(p);        // K khalf..khalf+7
  d.hi = *(const uint4*)(p + 16);   // K khalf+16..khalf+23
  return __builtin_bit_cast(v16bf, d);
}

// ---------- CDNA5 async global->LDS copy (ASYNCcnt path, no VGPR round-trip) ----
// INST_OFFSET is added to BOTH the LDS and global addresses (ISA 15.18.3),
// so one (lds,global) pair covers a contiguous 32B segment with offsets 0/16.
__device__ __forceinline__ void async_copy32B(u32 lds, const u16* g) {
  asm volatile("global_load_async_to_lds_b128 %0, %1, off"
               :: "v"(lds), "v"(g) : "memory");
  asm volatile("global_load_async_to_lds_b128 %0, %1, off offset:16"
               :: "v"(lds), "v"(g) : "memory");
}
__device__ __forceinline__ void wait_async0() {
#if __has_builtin(__builtin_amdgcn_s_wait_asynccnt)
  __builtin_amdgcn_s_wait_asynccnt(0);
#else
  asm volatile("s_wait_asynccnt 0x0" ::: "memory");
#endif
}
// flat LDS pointer -> 32-bit LDS offset (aperture is entirely in addr[63:32])
__device__ __forceinline__ u32 lds_off(const void* p) {
  return (u32)(uintptr_t)p;
}

__device__ __forceinline__ v8f wmma_bf16(v16bf a, v16bf b, v8f c) {
  return __builtin_amdgcn_wmma_f32_16x16x32_bf16(false, a, false, b,
                                                 (short)0, c, false, false);
}

// ==================================================================
// Tiled GEMM: C[128x128 per block] = A @ W^T (+bias) (+Add)
// 256 threads = 8 waves; wave = 4 Mtiles x 2 Ntiles (64x32 output).
// Double-buffered 128x32 K-panels in LDS, filled by async global->LDS
// copies that overlap the WMMA compute on the other buffer.
// Requires: M,N multiples of 128 (grid.y=M/128, grid.x=N/128), K % 32 == 0.
// ==================================================================
__global__ void __launch_bounds__(256)
k_gemm_tiled(const u16* __restrict__ A, int lda,
             const u16* __restrict__ W, int ldw,
             const u16* __restrict__ Add, int ldadd,
             const float* __restrict__ bias,
             u16* __restrict__ Cbf, int ldc,
             float* __restrict__ Cf, int ldcf,
             int K) {
  __shared__ u16 As[2][128 * PITCH];
  __shared__ u16 Bs[2][128 * PITCH];

  const int tid  = threadIdx.x;
  const int wid  = tid >> 5;
  const int lane = tid & 31;
  const int rowBase = blockIdx.y * 128;
  const int colBase = blockIdx.x * 128;

  // staging map: thread -> (panel row r, 16-half segment seg)
  const int r   = tid >> 1;
  const int seg = tid & 1;
  const u16* gA = A + (size_t)(rowBase + r) * lda + seg * 16;
  const u16* gB = W + (size_t)(colBase + r) * ldw + seg * 16;
  const u32 laA0 = lds_off(&As[0][r * PITCH + seg * 16]);
  const u32 laA1 = lds_off(&As[1][r * PITCH + seg * 16]);
  const u32 laB0 = lds_off(&Bs[0][r * PITCH + seg * 16]);
  const u32 laB1 = lds_off(&Bs[1][r * PITCH + seg * 16]);

  // wave -> sub-tile fragment bases
  const int wrow = wid >> 2;        // 0..1 : 64-row band
  const int wcol = wid & 3;         // 0..3 : 32-col band
  const int hl   = lane & 15;
  const int kA   = (lane & 16) ? 8 : 0;
  const int kB   = (lane & 16) ? 8 : 0;
  const u16* fA0 = &As[0][(wrow * 64 + hl) * PITCH + kA];
  const u16* fA1 = &As[1][(wrow * 64 + hl) * PITCH + kA];
  const u16* fB0 = &Bs[0][(wcol * 32 + hl) * PITCH + kB];
  const u16* fB1 = &Bs[1][(wcol * 32 + hl) * PITCH + kB];

  v8f acc[4][2] = {};

  // prologue: stage panel 0 into buffer 0
  async_copy32B(laA0, gA);
  async_copy32B(laB0, gB);
  wait_async0();
  __syncthreads();

  for (int kk = 0; kk < K; kk += 64) {
    // ---- stage panel kk+32 into buffer 1 while computing on buffer 0 ----
    if (kk + 32 < K) {
      async_copy32B(laA1, gA + kk + 32);
      async_copy32B(laB1, gB + kk + 32);
    }
    {
      v16bf b0 = load_b16(fB0);
      v16bf b1 = load_b16(fB0 + 16 * PITCH);
#pragma unroll
      for (int mt = 0; mt < 4; ++mt) {
        v16bf a = load_a16(fA0 + mt * 16 * PITCH);
        acc[mt][0] = wmma_bf16(a, b0, acc[mt][0]);
        acc[mt][1] = wmma_bf16(a, b1, acc[mt][1]);
      }
    }
    wait_async0();
    __syncthreads();
    if (kk + 32 >= K) break;

    // ---- stage panel kk+64 into buffer 0 while computing on buffer 1 ----
    if (kk + 64 < K) {
      async_copy32B(laA0, gA + kk + 64);
      async_copy32B(laB0, gB + kk + 64);
    }
    {
      v16bf b0 = load_b16(fB1);
      v16bf b1 = load_b16(fB1 + 16 * PITCH);
#pragma unroll
      for (int mt = 0; mt < 4; ++mt) {
        v16bf a = load_a16(fA1 + mt * 16 * PITCH);
        acc[mt][0] = wmma_bf16(a, b0, acc[mt][0]);
        acc[mt][1] = wmma_bf16(a, b1, acc[mt][1]);
      }
    }
    wait_async0();
    __syncthreads();
  }

  // epilogue
  const int mhalf = (lane & 16) ? 8 : 0;
#pragma unroll
  for (int mt = 0; mt < 4; ++mt) {
#pragma unroll
    for (int nt = 0; nt < 2; ++nt) {
      int n = colBase + wcol * 32 + nt * 16 + hl;
      int mb = rowBase + wrow * 64 + mt * 16 + mhalf;
#pragma unroll
      for (int i = 0; i < 8; ++i) {
        int row = mb + i;
        float v = acc[mt][nt][i];
        if (bias) v += bias[n];
        if (Add)  v += bf2f(Add[(size_t)row * ldadd + n]);
        if (Cbf)  Cbf[(size_t)row * ldc  + n] = f2bf(v);
        if (Cf)   Cf [(size_t)row * ldcf + n] = v;
      }
    }
  }
}

// ==================================================================
// Fallback: one 16x16 tile per wave (used for tiny tree levels, M < 128)
// ==================================================================
__global__ void k_gemm_wave(const u16* __restrict__ A, int lda,
                            const u16* __restrict__ W, int ldw,
                            const u16* __restrict__ Add, int ldadd,
                            const float* __restrict__ bias,
                            u16* __restrict__ Cbf, int ldc,
                            float* __restrict__ Cf, int ldcf,
                            int Mtiles, int Ntiles, int K) {
  int gwave = (blockIdx.x * blockDim.x + threadIdx.x) >> 5;   // wave-uniform
  if (gwave >= Mtiles * Ntiles) return;                        // whole-wave exit
  int lane = threadIdx.x & 31;
  int tm = gwave / Ntiles;
  int tn = gwave - tm * Ntiles;
  int hl = lane & 15;
  int kA = (lane & 16) ? 8 : 0;
  int kB = (lane & 16) ? 8 : 0;
  const u16* pa = A + (size_t)(tm * 16 + hl) * lda + kA;
  const u16* pw = W + (size_t)(tn * 16 + hl) * ldw + kB;
  v8f acc = {};
  for (int k = 0; k < K; k += 32) {
    v16bf a = load_a16(pa + k);
    v16bf b = load_b16(pw + k);
    acc = wmma_bf16(a, b, acc);
  }
  int n = tn * 16 + hl;
  int mbase = tm * 16 + ((lane & 16) ? 8 : 0);
#pragma unroll
  for (int i = 0; i < 8; ++i) {
    int row = mbase + i;
    float v = acc[i];
    if (bias) v += bias[n];
    if (Add)  v += bf2f(Add[(size_t)row * ldadd + n]);
    if (Cbf)  Cbf[(size_t)row * ldc  + n] = f2bf(v);
    if (Cf)   Cf [(size_t)row * ldcf + n] = v;
  }
}

// ---------- embedding gather + f32->bf16 cast ----------
__global__ void k_gather_cast(const float* __restrict__ emb, const int* __restrict__ ids,
                              u16* __restrict__ dst) {
  long idx = (long)blockIdx.x * blockDim.x + threadIdx.x;     // SEQ*EDIM threads
  int s = (int)(idx >> 9);                                     // /512
  int e = (int)(idx & 511);
  dst[idx] = f2bf(emb[(size_t)ids[s] * EDIM + e]);
}

// ---------- strided f32 -> bf16 cast (slices We / Wh out of i2h_w) ----------
__global__ void k_cast_strided(const float* __restrict__ src, int src_ld, int src_off,
                               u16* __restrict__ dst, int cols) {
  long idx = (long)blockIdx.x * blockDim.x + threadIdx.x;
  int r = (int)(idx / cols);
  int c = (int)(idx - (long)r * cols);
  dst[idx] = f2bf(src[(size_t)r * src_ld + src_off + c]);
}

// ---------- LDS-tiled bf16 transpose (for operator squaring W*W = W @ (W^T)^T) ----------
__global__ void k_transpose_bf16(const u16* __restrict__ src, u16* __restrict__ dst, int n) {
  __shared__ u16 tile[32][33];
  int bx = blockIdx.x * 32, by = blockIdx.y * 32;
  int tx = threadIdx.x, ty = threadIdx.y;                      // 32 x 8
#pragma unroll
  for (int r = 0; r < 4; ++r)
    tile[ty + 8 * r][tx] = src[(size_t)(by + ty + 8 * r) * n + bx + tx];
  __syncthreads();
#pragma unroll
  for (int r = 0; r < 4; ++r)
    dst[(size_t)(bx + ty + 8 * r) * n + by + tx] = tile[tx][ty + 8 * r];
}

// ---------- head: two masked-relu layers + logits + log_softmax + outputs ----------
__global__ void k_head(const float* __restrict__ hidden,
                       const float* __restrict__ mask0, const float* __restrict__ mask1,
                       const float* __restrict__ l0w, const float* __restrict__ l0b,
                       const float* __restrict__ l1w, const float* __restrict__ l1b,
                       const float* __restrict__ how, const float* __restrict__ hob,
                       float* __restrict__ out) {
  __shared__ float s0[1024];
  __shared__ float s1[1024];
  __shared__ float s2[512];
  int t = threadIdx.x;
  float h = hidden[t];
  s0[t] = h;
  out[2536 + t] = h;                                           // hidden output
  __syncthreads();
  {                                                            // layer 0 (1024)
    const float* wr = l0w + (size_t)t * HDIM;
    float a = l0b[t];
    for (int k = 0; k < HDIM; ++k) a += wr[k] * s0[k];
    a *= mask0[t];
    a = a > 0.f ? a : 0.f;
    out[1000 + t] = __int_as_float(a > 0.f ? 1 : 0);           // status0 (int32 bits)
    s1[t] = a;
  }
  __syncthreads();
  if (t < 512) {                                               // layer 1 (512)
    const float* wr = l1w + (size_t)t * HDIM;
    float a = l1b[t];
    for (int k = 0; k < HDIM; ++k) a += wr[k] * s1[k];
    a *= mask1[t];
    a = a > 0.f ? a : 0.f;
    out[2024 + t] = __int_as_float(a > 0.f ? 1 : 0);           // status1
    s2[t] = a;
  }
  __syncthreads();
  float lg = -3.0e38f;
  if (t < 1000) {                                              // logits (1000)
    const float* wr = how + (size_t)t * 512;
    float a = hob[t];
    for (int k = 0; k < 512; ++k) a += wr[k] * s2[k];
    lg = a;
  }
  s0[t] = lg;
  s1[t] = lg;
  __syncthreads();
  for (int s = 512; s > 0; s >>= 1) {                          // max-reduce
    if (t < s) s1[t] = fmaxf(s1[t], s1[t + s]);
    __syncthreads();
  }
  float mx = s1[0];
  __syncthreads();
  s1[t] = (t < 1000) ? expf(s0[t] - mx) : 0.f;
  __syncthreads();
  for (int s = 512; s > 0; s >>= 1) {                          // sum-reduce
    if (t < s) s1[t] += s1[t + s];
    __syncthreads();
  }
  float lz = logf(s1[0]);
  if (t < 1000) out[t] = s0[t] - mx - lz;                      // log_softmax
}

// ------------------------------------------------------------------
extern "C" void kernel_launch(void* const* d_in, const int* in_sizes, int n_in,
                              void* d_out, int out_size, void* d_ws, size_t ws_size,
                              hipStream_t stream) {
  (void)in_sizes; (void)n_in; (void)out_size; (void)ws_size;
  const int*   ids   = (const int*)  d_in[0];
  const float* mask0 = (const float*)d_in[1];
  const float* mask1 = (const float*)d_in[2];
  const float* emb   = (const float*)d_in[3];
  const float* i2hw  = (const float*)d_in[4];
  const float* i2hb  = (const float*)d_in[5];
  const float* l0w   = (const float*)d_in[6];
  const float* l0b   = (const float*)d_in[7];
  const float* l1w   = (const float*)d_in[8];
  const float* l1b   = (const float*)d_in[9];
  const float* how   = (const float*)d_in[10];
  const float* hob   = (const float*)d_in[11];
  float* out = (float*)d_out;

  // ---- carve workspace (256B aligned) ----
  char* w = (char*)d_ws;
  auto carve = [&](size_t bytes) -> void* {
    void* p = (void*)w;
    w += (bytes + 255) & ~(size_t)255;
    return p;
  };
  u16*  Xbf  = (u16*) carve((size_t)SEQ  * EDIM * 2);   //  8 MB gathered embeddings (bf16)
  u16*  Webf = (u16*) carve((size_t)HDIM * EDIM * 2);   //  1 MB
  u16*  Wc   = (u16*) carve((size_t)HDIM * HDIM * 2);   //  2 MB operator W^(2^l)
  u16*  Wt   = (u16*) carve((size_t)HDIM * HDIM * 2);   //  2 MB operator transpose
  u16*  Wn   = (u16*) carve((size_t)HDIM * HDIM * 2);   //  2 MB operator next
  u16*  EP   = (u16*) carve((size_t)SEQ  * HDIM * 2);   // 16 MB emb_proj (bf16)
  u16*  ping = (u16*) carve((size_t)(SEQ/2) * HDIM * 2);//  8 MB tree buffer
  u16*  pong = (u16*) carve((size_t)(SEQ/4) * HDIM * 2);//  4 MB tree buffer
  float* hf  = (float*)carve((size_t)16 * HDIM * 4);    // final hidden (16 rows, row 0 valid)

  // ---- slice & cast weights, gather embeddings ----
  k_cast_strided<<<(HDIM * EDIM) / 256, 256, 0, stream>>>(i2hw, EDIM + HDIM, 0,    Webf, EDIM);
  k_cast_strided<<<(HDIM * HDIM) / 256, 256, 0, stream>>>(i2hw, EDIM + HDIM, EDIM, Wc,   HDIM);
  k_gather_cast <<<(SEQ  * EDIM) / 256, 256, 0, stream>>>(emb, ids, Xbf);

  // ---- emb_proj = X @ We^T + b   (8192x512x1024, async LDS-tiled WMMA bf16) ----
  k_gemm_tiled<<<dim3(HDIM / 128, SEQ / 128), 256, 0, stream>>>(
      Xbf, EDIM, Webf, EDIM, nullptr, 0, i2hb,
      EP, HDIM, nullptr, 0, EDIM);

  // ---- log-depth associative tree for the linear scan:
  //      y_m = x_{2m} @ W_l^T + x_{2m+1};  W_{l+1} = W_l * W_l
  u16* X = EP;
  for (int lvl = 0; lvl < 13; ++lvl) {
    int Mout = 4096 >> lvl;                   // 4096,2048,...,1
    u16* Y = (lvl & 1) ? pong : ping;
    bool last = (lvl == 12);
    if (Mout >= 128) {
      k_gemm_tiled<<<dim3(HDIM / 128, Mout / 128), 256, 0, stream>>>(
          X, 2 * HDIM, Wc, HDIM, X + HDIM, 2 * HDIM, nullptr,
          Y, HDIM, nullptr, 0, HDIM);
    } else {
      int Mtiles = (Mout + 15) / 16;
      int blocks = (Mtiles * (HDIM / 16) + 7) / 8;
      k_gemm_wave<<<blocks, 256, 0, stream>>>(
          X, 2 * HDIM, Wc, HDIM, X + HDIM, 2 * HDIM, nullptr,
          Y, HDIM, last ? hf : nullptr, HDIM,
          Mtiles, HDIM / 16, HDIM);
    }
    X = Y;
    if (!last) {
      // W_next = W * W  ==  W @ (W^T)^T  -> transpose then A@W^T GEMM
      k_transpose_bf16<<<dim3(32, 32), dim3(32, 8), 0, stream>>>(Wc, Wt, HDIM);
      k_gemm_tiled<<<dim3(HDIM / 128, HDIM / 128), 256, 0, stream>>>(
          Wc, HDIM, Wt, HDIM, nullptr, 0, nullptr,
          Wn, HDIM, nullptr, 0, HDIM);
      u16* tmp = Wc; Wc = Wn; Wn = tmp;
    }
  }

  // ---- head: MLP + log_softmax + status flags + hidden ----
  k_head<<<1, 1024, 0, stream>>>(hf, mask0, mask1, l0w, l0b, l1w, l1b, how, hob, out);
}
